// GIN_35485019799983
// MI455X (gfx1250) — matrix operand order repaired and checked by reading, not compiled, not run on hardware
//
#include <hip/hip_runtime.h>
#include <hip/hip_bf16.h>
#include <math.h>

// ---------------------------------------------------------------------------
// GIN forward on MI455X (gfx1250, wave32).
//  - GEMMs via v_wmma_f32_16x16x32_bf16 (bf16 inputs, f32 accumulate).
//  - A tiles staged to LDS with GLOBAL_LOAD_ASYNC_TO_LDS_B128 (ASYNCcnt),
//    double-buffered; fragments read from LDS (ds_load_b128), coalesced
//    global traffic instead of 16-line gathers.
//  - Weights pre-transposed to bf16 [N][K]: B fragments are contiguous 16B
//    loads, L1/L2 resident (128KB total).
//  - Scatter-add: one wave per edge, coalesced global_atomic_add_f32.
//  - Working set (~150MB) fits in the 192MB L2.
// ---------------------------------------------------------------------------

typedef __attribute__((ext_vector_type(16))) __bf16 v16bf;
typedef __attribute__((ext_vector_type(8)))  float  v8f;

union BF16x16 { v16bf v; unsigned int u[8]; unsigned short s[16]; };

__device__ __forceinline__ unsigned short f2bf(float f) {
    union { float f; unsigned int u; } x; x.f = f;
    unsigned int r = x.u + 0x7FFFu + ((x.u >> 16) & 1u);  // RNE
    return (unsigned short)(r >> 16);
}
__device__ __forceinline__ unsigned int pk2bf(float lo, float hi) {
    return (unsigned int)f2bf(lo) | ((unsigned int)f2bf(hi) << 16);
}

// async copy of 16B from global to LDS; tracked by ASYNCcnt
__device__ __forceinline__ void async_b128_to_lds(unsigned lds_off, const float* g) {
    asm volatile("global_load_async_to_lds_b128 %0, %1, off"
                 :: "v"(lds_off), "v"(g) : "memory");
}

#define S_WAIT_ASYNC_LE4() asm volatile("s_wait_asynccnt 0x4" ::: "memory")
#define S_WAIT_ASYNC_0()   asm volatile("s_wait_asynccnt 0x0" ::: "memory")

// ---------------------------------------------------------------------------
// C[M,N] = act(A[M,K] @ W[K,N] + bias), W given transposed bf16 [N][K].
// Block = 128 threads = 4 waves; block tile 64 rows x NT*16 cols.
// A tile (64 x 32 f32) double-buffered in LDS via async DMA.
// Fragment layouts (ISA 7.12.2, wave32):
//   A 16x32 bf16: lane%16 = row M; lane/16 selects K-half:
//     elems 0..7  = K 8*half .. +7,  elems 8..15 = K 16+8*half .. +7
//   B 32x16 bf16: lane%16 = col N; same K order as A
//   C 16x16 f32:  elem r -> row r+8*(lane/16), col lane%16
// ---------------------------------------------------------------------------
#define LDSROW 36   // padded row stride (floats), 16B-aligned, spreads banks

template<int NT>
__global__ __launch_bounds__(128) void gemm_bf16_wmma(
    const float* __restrict__ A,
    const unsigned short* __restrict__ Wt, const float* __restrict__ bias,
    float* __restrict__ C, int M, int K, int N, int relu)
{
    __shared__ float lds_a[2][64 * LDSROW];

    const int tid  = threadIdx.x;
    const int wave = tid >> 5;
    const int lane = tid & 31;
    const int half = lane >> 4;
    const int lm   = lane & 15;

    const int mblk  = blockIdx.y * 64;
    const int nbase = blockIdx.x * (NT * 16);

    const unsigned lbase0 = (unsigned)(size_t)(&lds_a[0][0]);
    const unsigned lbase1 = (unsigned)(size_t)(&lds_a[1][0]);

    // staging: thread covers 4 x b128; f = tid + 128*i -> row = tid/8 + 16i,
    // col4 = tid&7 (4-float column group). Coalesced: 8 lanes span a row seg.
    const int col4 = tid & 7;
    const int row0 = tid >> 3;

    auto stage = [&](int buf, int k0) {
        const unsigned lb = buf ? lbase1 : lbase0;
#pragma unroll
        for (int i = 0; i < 4; ++i) {
            const int row  = row0 + 16 * i;
            int grow = mblk + row; if (grow >= M) grow = M - 1;  // clamp tail
            const float* g = A + (size_t)grow * K + k0 + col4 * 4;
            const unsigned loff = lb + (unsigned)((row * LDSROW + col4 * 4) * 4);
            async_b128_to_lds(loff, g);
        }
    };

    const unsigned short* Bp[NT];
    float biasv[NT];
#pragma unroll
    for (int t = 0; t < NT; ++t) {
        const int n = nbase + t * 16 + lm;
        Bp[t]    = Wt + (size_t)n * K + half * 8;
        biasv[t] = bias[n];
    }

    v8f acc[NT];
    const v8f zv = {0.f, 0.f, 0.f, 0.f, 0.f, 0.f, 0.f, 0.f};
#pragma unroll
    for (int t = 0; t < NT; ++t) acc[t] = zv;

    const int lr     = wave * 16 + lm;   // my A-fragment row within the tile
    const int nsteps = K >> 5;

    stage(0, 0);                         // prologue DMA for step 0
    for (int s = 0; s < nsteps; ++s) {
        const int  cur  = s & 1;
        const bool more = (s + 1) < nsteps;
        if (more) {
            stage(1 - cur, (s + 1) << 5);   // overlap next tile's DMA
            S_WAIT_ASYNC_LE4();             // wait only for current tile
        } else {
            S_WAIT_ASYNC_0();
        }
        __syncthreads();                    // current tile visible to all

        // A fragment from LDS (ds_load_b128 x4), then f32 -> bf16
        const int abase = lr * LDSROW + half * 8;
        const float4 a0 = *(const float4*)&lds_a[cur][abase];
        const float4 a1 = *(const float4*)&lds_a[cur][abase + 4];
        const float4 a2 = *(const float4*)&lds_a[cur][abase + 16];
        const float4 a3 = *(const float4*)&lds_a[cur][abase + 20];
        BF16x16 af;
        af.u[0] = pk2bf(a0.x, a0.y); af.u[1] = pk2bf(a0.z, a0.w);
        af.u[2] = pk2bf(a1.x, a1.y); af.u[3] = pk2bf(a1.z, a1.w);
        af.u[4] = pk2bf(a2.x, a2.y); af.u[5] = pk2bf(a2.z, a2.w);
        af.u[6] = pk2bf(a3.x, a3.y); af.u[7] = pk2bf(a3.z, a3.w);

        const int k0 = s << 5;
#pragma unroll
        for (int t = 0; t < NT; ++t) {
            BF16x16 bf;
            const uint4 b01 = *(const uint4*)(Bp[t] + k0);      // K k0+8h..+7
            const uint4 b23 = *(const uint4*)(Bp[t] + k0 + 16); // K k0+16+8h..+7
            bf.u[0] = b01.x; bf.u[1] = b01.y; bf.u[2] = b01.z; bf.u[3] = b01.w;
            bf.u[4] = b23.x; bf.u[5] = b23.y; bf.u[6] = b23.z; bf.u[7] = b23.w;
            acc[t] = __builtin_amdgcn_wmma_f32_16x16x32_bf16(
                false, af.v, false, bf.v, (short)0, acc[t], false, false);
        }
        __syncthreads();   // everyone done with buf[cur] before it's re-staged
    }

#pragma unroll
    for (int t = 0; t < NT; ++t) {
        const int n = nbase + t * 16 + lm;
#pragma unroll
        for (int r = 0; r < 8; ++r) {
            const int row = mblk + wave * 16 + r + 8 * half;
            if (row < M) {
                float v = acc[t][r] + biasv[t];
                if (relu) v = fmaxf(v, 0.0f);
                C[(size_t)row * N + n] = v;
            }
        }
    }
}

// agg[dst] += h[src] : one wave per edge, lanes stride the 256 features.
__global__ __launch_bounds__(256) void gin_scatter_add(
    const float* __restrict__ h, const int* __restrict__ src,
    const int* __restrict__ dst, float* __restrict__ agg, int nE)
{
    const int wid  = (int)((blockIdx.x * (unsigned)blockDim.x + threadIdx.x) >> 5);
    const int lane = threadIdx.x & 31;
    if (wid >= nE) return;
    const int s = src[wid];
    const int d = dst[wid];
    const float* hs = h   + (size_t)s * 256;
    float*       ad = agg + (size_t)d * 256;
#pragma unroll
    for (int i = 0; i < 8; ++i) {
        const int f = lane + 32 * i;            // coalesced atomics across wave
        atomicAdd(ad + f, hs[f]);               // global_atomic_add_f32
    }
}

// dst[n*K + k] = bf16(src[k*N + n])   (weights: [K,N] f32 -> [N][K] bf16)
__global__ void transpose_cvt_bf16(const float* __restrict__ src,
                                   unsigned short* __restrict__ dst, int K, int N)
{
    int t = blockIdx.x * blockDim.x + threadIdx.x;
    if (t >= K * N) return;
    int k = t / N, n = t - k * N;
    dst[n * K + k] = f2bf(src[t]);
}

// log_softmax over rows of [M,16]
__global__ __launch_bounds__(256) void logsoftmax16(
    const float* __restrict__ L, float* __restrict__ out, int M)
{
    int r = blockIdx.x * blockDim.x + threadIdx.x;
    if (r >= M) return;
    const float* p = L + (size_t)r * 16;
    float v[16];
    float4 q0 = *(const float4*)(p + 0);
    float4 q1 = *(const float4*)(p + 4);
    float4 q2 = *(const float4*)(p + 8);
    float4 q3 = *(const float4*)(p + 12);
    v[0]=q0.x; v[1]=q0.y; v[2]=q0.z; v[3]=q0.w;
    v[4]=q1.x; v[5]=q1.y; v[6]=q1.z; v[7]=q1.w;
    v[8]=q2.x; v[9]=q2.y; v[10]=q2.z; v[11]=q2.w;
    v[12]=q3.x; v[13]=q3.y; v[14]=q3.z; v[15]=q3.w;
    float mx = v[0];
#pragma unroll
    for (int i = 1; i < 16; ++i) mx = fmaxf(mx, v[i]);
    float s = 0.f;
#pragma unroll
    for (int i = 0; i < 16; ++i) s += __expf(v[i] - mx);
    const float ls = mx + __logf(s);
    float* o = out + (size_t)r * 16;
    *(float4*)(o + 0)  = make_float4(v[0]-ls,  v[1]-ls,  v[2]-ls,  v[3]-ls);
    *(float4*)(o + 4)  = make_float4(v[4]-ls,  v[5]-ls,  v[6]-ls,  v[7]-ls);
    *(float4*)(o + 8)  = make_float4(v[8]-ls,  v[9]-ls,  v[10]-ls, v[11]-ls);
    *(float4*)(o + 12) = make_float4(v[12]-ls, v[13]-ls, v[14]-ls, v[15]-ls);
}

extern "C" void kernel_launch(void* const* d_in, const int* in_sizes, int n_in,
                              void* d_out, int out_size, void* d_ws, size_t ws_size,
                              hipStream_t stream)
{
    const float* x     = (const float*)d_in[0];
    const int*   eidx  = (const int*)d_in[1];   // [2, E] int32
    const float* pre_w = (const float*)d_in[2];
    const float* pre_b = (const float*)d_in[3];
    const float* w1    = (const float*)d_in[4]; // [3,256,256]
    const float* b1    = (const float*)d_in[5]; // [3,256]
    const float* w2    = (const float*)d_in[6];
    const float* b2    = (const float*)d_in[7];
    const float* postw = (const float*)d_in[8];
    const float* postb = (const float*)d_in[9];
    const float* readw = (const float*)d_in[10]; // [256,16]
    const float* readb = (const float*)d_in[11];

    const int Nn = 50000, E = 800000, F = 128, H = 256, Cc = 16, Ly = 3;
    const int* src = eidx;
    const int* dst = eidx + E;

    char* ws = (char*)d_ws;
    size_t off = 0;
    auto walloc = [&](size_t bytes) -> void* {
        void* p = ws + off;
        off += (bytes + 255) & ~(size_t)255;
        return p;
    };
    float* h      = (float*)walloc((size_t)Nn * H * 4);
    float* agg    = (float*)walloc((size_t)Nn * H * 4);
    float* z      = (float*)walloc((size_t)Nn * H * 4);
    float* logits = (float*)walloc((size_t)Nn * Cc * 4);
    unsigned short* WtPre  = (unsigned short*)walloc((size_t)H * F * 2);
    unsigned short* Wt1[3], *Wt2[3];
    for (int l = 0; l < Ly; ++l) {
        Wt1[l] = (unsigned short*)walloc((size_t)H * H * 2);
        Wt2[l] = (unsigned short*)walloc((size_t)H * H * 2);
    }
    unsigned short* WtPost = (unsigned short*)walloc((size_t)H * H * 2);
    unsigned short* WtRead = (unsigned short*)walloc((size_t)Cc * H * 2);

    // ---- weight prep: transpose + f32->bf16 ----
    auto tl = [&](const float* s, unsigned short* d, int K, int N) {
        int n = K * N;
        transpose_cvt_bf16<<<(n + 255) / 256, 256, 0, stream>>>(s, d, K, N);
    };
    tl(pre_w, WtPre, F, H);
    for (int l = 0; l < Ly; ++l) {
        tl(w1 + (size_t)l * H * H, Wt1[l], H, H);
        tl(w2 + (size_t)l * H * H, Wt2[l], H, H);
    }
    tl(postw, WtPost, H, H);
    tl(readw, WtRead, H, Cc);

    const dim3 blk(128);
    const int mg = (Nn + 63) / 64;     // 782 row-blocks

    // pre: h = x @ pre_w + pre_b   (no relu)
    gemm_bf16_wmma<4><<<dim3(H / 64, mg), blk, 0, stream>>>(
        x, WtPre, pre_b, h, Nn, F, H, 0);

    for (int l = 0; l < Ly; ++l) {
        // agg = h, then agg[dst] += h[src]  ->  agg = h + sum_{j->i} h_j
        hipMemcpyAsync(agg, h, (size_t)Nn * H * 4, hipMemcpyDeviceToDevice, stream);
        gin_scatter_add<<<E / 8, 256, 0, stream>>>(h, src, dst, agg, E);
        // z = relu(agg @ w1 + b1)
        gemm_bf16_wmma<4><<<dim3(H / 64, mg), blk, 0, stream>>>(
            agg, Wt1[l], b1 + (size_t)l * H, z, Nn, H, H, 1);
        // h = relu(z @ w2 + b2)
        gemm_bf16_wmma<4><<<dim3(H / 64, mg), blk, 0, stream>>>(
            z, Wt2[l], b2 + (size_t)l * H, h, Nn, H, H, 1);
    }

    // p = relu(h @ post_w + post_b)  -> reuse z
    gemm_bf16_wmma<4><<<dim3(H / 64, mg), blk, 0, stream>>>(
        h, WtPost, postb, z, Nn, H, H, 1);
    // logits = p @ read_w + read_b   (N=16 -> single tile strip per wave)
    gemm_bf16_wmma<1><<<dim3(1, mg), blk, 0, stream>>>(
        z, WtRead, readb, logits, Nn, H, Cc, 0);

    // logp -> d_out[0 .. Nn*16)
    logsoftmax16<<<(Nn + 255) / 256, 256, 0, stream>>>(logits, (float*)d_out, Nn);

    // x1 (= h after conv stack) duplicated into outputs 2 and 3
    float* outp = (float*)d_out;
    hipMemcpyAsync(outp + (size_t)Nn * Cc, h, (size_t)Nn * H * 4,
                   hipMemcpyDeviceToDevice, stream);
    hipMemcpyAsync(outp + (size_t)Nn * Cc + (size_t)Nn * H, h, (size_t)Nn * H * 4,
                   hipMemcpyDeviceToDevice, stream);
}
